// Block_90185723281472
// MI455X (gfx1250) — matrix operand order repaired and checked by reading, not compiled, not run on hardware
//
#include <hip/hip_runtime.h>
#include <hip/hip_bf16.h>
#include <cstdint>

typedef _Float16 half_t;
typedef __attribute__((ext_vector_type(16))) _Float16 v16h;
typedef __attribute__((ext_vector_type(8)))  float    v8f;
typedef __attribute__((ext_vector_type(4)))  float    v4f;
typedef __attribute__((ext_vector_type(4)))  unsigned int v4u;
typedef __attribute__((ext_vector_type(8)))  int      v8i;
typedef __attribute__((ext_vector_type(4)))  int      v4i;

#if __has_builtin(__builtin_amdgcn_tensor_load_to_lds)
#define HAS_TDM 1
#if defined(__has_include)
#if __has_include(<hip/amd_detail/amd_gfx1250_TDM.h>)
#define TDM_6ARG 1
#endif
#endif
#endif

__device__ inline v8f wmma32(v16h a, v16h b, v8f c) {
  // (neg_a, A, neg_b, B, c_mod, C, reuse_a, reuse_b)
  return __builtin_amdgcn_wmma_f32_16x16x32_f16(false, a, false, b, (short)0, c, false, false);
}

// A-operand fragment (16x32 f16): lane (r, h) holds K = h*16 + 0..15 contiguous.
__device__ inline v16h ld_a(const half_t* p) {
  union { v4f q[2]; v16h v; } u;
  u.q[0] = *reinterpret_cast<const v4f*>(p);
  u.q[1] = *reinterpret_cast<const v4f*>(p + 8);
  return u.v;
}

// B-operand fragment (32x16 f16): lane (c, h) holds K = h*16 + 0..15 contiguous.
__device__ inline v16h ld_b(const half_t* p) {
  union { v4f q[2]; v16h v; } u;
  u.q[0] = *reinterpret_cast<const v4f*>(p);
  u.q[1] = *reinterpret_cast<const v4f*>(p + 8);
  return u.v;
}

__device__ inline float gelu_f(float x) {
  return 0.5f * x * (1.f + erff(x * 0.70710678118654752f));
}

#ifdef HAS_TDM
// TDM: async-load a [tile_h x tile_w] f16 tile (row stride = row_stride elems)
// from global into LDS at lds_off, with LDS padding of 16B after every 128B
// (tile_w must be 64) so staged rows have a 144B stride.
__device__ inline void tdm_load_tile_f16(unsigned lds_off, const void* gaddr,
                                         int tensor_w, int tensor_h,
                                         int tile_w, int tile_h, int row_stride) {
  uint64_t ga = (uint64_t)(uintptr_t)gaddr;
  v4u g0 = { 1u,                                   // count = 1 valid descriptor
             lds_off,                              // lds_addr
             (unsigned)ga,                         // global_addr[31:0]
             (unsigned)(ga >> 32) | (2u << 30) };  // global_addr[56:32] | type=2
  // group1: data_size=2B (code 1), pad_enable, pad_interval=32 DW, pad_amount=4 DW
  unsigned w0 = (1u << 16) | (1u << 20) | (4u << 22) | (3u << 25);
  unsigned w1 = ((unsigned)tensor_w & 0xffffu) << 16;                      // tensor_dim0 lo
  unsigned w2 = (((unsigned)tensor_w >> 16) & 0xffffu) |
                (((unsigned)tensor_h & 0xffffu) << 16);                    // dim0 hi | dim1 lo
  unsigned w3 = (((unsigned)tensor_h >> 16) & 0xffffu) |
                (((unsigned)tile_w & 0xffffu) << 16);                      // dim1 hi | tile_dim0
  unsigned w4 = ((unsigned)tile_h & 0xffffu);                              // tile_dim1 | tile_dim2=0
  unsigned w5 = (unsigned)row_stride;                                      // tensor_dim0_stride lo32
  v8i g1 = { (int)w0, (int)w1, (int)w2, (int)w3, (int)w4, (int)w5, 0, 0 };
  v4i gz = { 0, 0, 0, 0 };
#ifdef TDM_6ARG
  v8i gz8 = { 0, 0, 0, 0, 0, 0, 0, 0 };
  __builtin_amdgcn_tensor_load_to_lds(g0, g1, gz, gz, gz8, 0);
#else
  __builtin_amdgcn_tensor_load_to_lds(g0, g1, gz, gz, 0);
#endif
}
#endif

// ---------------------------------------------------------------------------
// fp32 [K][N] -> f16 transposed [N][K]
__global__ __launch_bounds__(256) void wcvt_kernel(const float* __restrict__ w,
                                                   half_t* __restrict__ wt,
                                                   int K, int N) {
  int idx = blockIdx.x * 256 + threadIdx.x;
  if (idx >= K * N) return;
  int n = idx / K, k = idx % K;
  wt[idx] = (half_t)w[(size_t)k * N + n];
}

// ---------------------------------------------------------------------------
// LayerNorm over 768 channels, fp32 in -> f16 out. One block per token.
__global__ __launch_bounds__(256) void ln_kernel(const float* __restrict__ x,
                                                 const float* __restrict__ w,
                                                 const float* __restrict__ bias,
                                                 half_t* __restrict__ out) {
  __shared__ float sb[256];
  int m = blockIdx.x, tid = threadIdx.x;
  const float* xr = x + (size_t)m * 768;
  float s = 0.f;
  for (int i = tid; i < 768; i += 256) s += xr[i];
  sb[tid] = s; __syncthreads();
  for (int st = 128; st; st >>= 1) { if (tid < st) sb[tid] += sb[tid + st]; __syncthreads(); }
  float mu = sb[0] * (1.f / 768.f);
  __syncthreads();
  float v = 0.f;
  for (int i = tid; i < 768; i += 256) { float d = xr[i] - mu; v += d * d; }
  sb[tid] = v; __syncthreads();
  for (int st = 128; st; st >>= 1) { if (tid < st) sb[tid] += sb[tid + st]; __syncthreads(); }
  float rstd = rsqrtf(sb[0] * (1.f / 768.f) + 1e-5f);
  half_t* orow = out + (size_t)m * 768;
  for (int i = tid; i < 768; i += 256)
    orow[i] = (half_t)((xr[i] - mu) * rstd * w[i] + bias[i]);
}

// ---------------------------------------------------------------------------
// WMMA GEMM: C[M,N] = A[M,K] (f16 row-major) x Bt[N,K]^T (f16, W transposed).
// 4 waves/block; wave = 16x64 strip; block tile 64x64; K chunked by 64 with the
// B-tile staged in LDS (TDM async DMA, double buffered; padded 144B row stride).
// mode 0: outh = acc (+bias)   mode 1: outh = gelu(acc+bias)
// mode 2: outf = acc + bias + resid
#define BK 64
#define BSTRIDE 72  /* 64 halves + 8 halves TDM padding */
__global__ __launch_bounds__(128) void gemm_kernel(
    const half_t* __restrict__ A, const half_t* __restrict__ Bt,
    const float* __restrict__ bias, const float* __restrict__ resid,
    half_t* __restrict__ outh, float* __restrict__ outf,
    int M, int N, int K, int mode) {
  __shared__ alignas(16) half_t Bs[2][64][BSTRIDE];
  int tid = threadIdx.x;
  int wave = tid >> 5, lane = tid & 31;
  int r = lane & 15, h = lane >> 4;
  int m0 = blockIdx.y * 64 + wave * 16;
  int n0 = blockIdx.x * 64;
  int arow = min(m0 + r, M - 1);
  const half_t* ap = A + (size_t)arow * K + h * 16;
  const half_t* btile = Bt + (size_t)n0 * K;  // 64 rows of length K
  int nk = K >> 6;
  v8f acc0 = {}, acc1 = {}, acc2 = {}, acc3 = {};

#ifdef HAS_TDM
  if (tid < 32)
    tdm_load_tile_f16((unsigned)(uintptr_t)&Bs[0][0][0], btile, K, 64, BK, 64, K);
  for (int i = 0; i < nk; ++i) {
    if (tid < 32) {
      if (i + 1 < nk) {
        tdm_load_tile_f16((unsigned)(uintptr_t)&Bs[(i + 1) & 1][0][0],
                          btile + (i + 1) * BK, K, 64, BK, 64, K);
        __builtin_amdgcn_s_wait_tensorcnt(1);
      } else {
        __builtin_amdgcn_s_wait_tensorcnt(0);
      }
    }
    __syncthreads();
    if (i + 1 < nk) __builtin_prefetch(ap + (i + 1) * BK, 0, 0);
    const half_t* bs = &Bs[i & 1][0][0];
    int kb = i * BK;
    for (int kk = 0; kk < BK; kk += 32) {
      v16h a = ld_a(ap + kb + kk);
      acc0 = wmma32(a, ld_b(bs + (r +  0) * BSTRIDE + kk + h * 16), acc0);
      acc1 = wmma32(a, ld_b(bs + (r + 16) * BSTRIDE + kk + h * 16), acc1);
      acc2 = wmma32(a, ld_b(bs + (r + 32) * BSTRIDE + kk + h * 16), acc2);
      acc3 = wmma32(a, ld_b(bs + (r + 48) * BSTRIDE + kk + h * 16), acc3);
    }
    __syncthreads();
  }
#else
  // Fallback: cooperative global->LDS staging, single barrier per chunk.
  for (int t = tid; t < 512; t += 128) {
    int row = t >> 3, c8 = (t & 7) * 8;
    *(v4f*)&Bs[0][row][c8] = *(const v4f*)(btile + (size_t)row * K + c8);
  }
  __syncthreads();
  for (int i = 0; i < nk; ++i) {
    if (i + 1 < nk) {
      const half_t* src = btile + (i + 1) * BK;
      for (int t = tid; t < 512; t += 128) {
        int row = t >> 3, c8 = (t & 7) * 8;
        *(v4f*)&Bs[(i + 1) & 1][row][c8] = *(const v4f*)(src + (size_t)row * K + c8);
      }
    }
    const half_t* bs = &Bs[i & 1][0][0];
    int kb = i * BK;
    for (int kk = 0; kk < BK; kk += 32) {
      v16h a = ld_a(ap + kb + kk);
      acc0 = wmma32(a, ld_b(bs + (r +  0) * BSTRIDE + kk + h * 16), acc0);
      acc1 = wmma32(a, ld_b(bs + (r + 16) * BSTRIDE + kk + h * 16), acc1);
      acc2 = wmma32(a, ld_b(bs + (r + 32) * BSTRIDE + kk + h * 16), acc2);
      acc3 = wmma32(a, ld_b(bs + (r + 48) * BSTRIDE + kk + h * 16), acc3);
    }
    __syncthreads();
  }
#endif

  v8f accs[4] = {acc0, acc1, acc2, acc3};
  for (int jj = 0; jj < 4; ++jj) {
    int n = n0 + jj * 16 + r;
    float bv = bias ? bias[n] : 0.f;
    union { float f[8]; v8f v; } cu; cu.v = accs[jj];
    for (int e = 0; e < 8; ++e) {
      int m = m0 + e + 8 * h;  // C layout: VGPR e -> row e (+8 for hi half-wave)
      if (m >= M) continue;
      float v = cu.f[e] + bv;
      if (mode == 1) v = gelu_f(v);
      size_t idx = (size_t)m * N + n;
      if (mode == 2) outf[idx] = v + resid[idx];
      else           outh[idx] = (half_t)v;
    }
  }
}

// ---------------------------------------------------------------------------
// Attention: one block = (batch b, head hd, 16-row q tile). N=1025 padded to 1056.
#define NCOL 1056
__global__ __launch_bounds__(128) void attn_kernel(const half_t* __restrict__ qkv,
                                                   half_t* __restrict__ o) {
  __shared__ alignas(16) half_t S[16][NCOL];
  __shared__ float red[16][8];
  __shared__ float rowstat[16];
  int tid = threadIdx.x;
  int wave = tid >> 5, lane = tid & 31;
  int r = lane & 15, h = lane >> 4;
  int qt = blockIdx.x, hd = blockIdx.y, b = blockIdx.z;
  int q0 = qt * 16;
  const size_t TS = 2304;
  const half_t* qb  = qkv + (size_t)b * 1025 * TS + hd * 64;
  const half_t* kb_ = qb + 768;
  const half_t* vb  = qb + 1536;

  for (int i = tid; i < 256; i += 128)
    S[i >> 4][1040 + (i & 15)] = (half_t)(-60000.f);

  // Phase 1: S = (Q K^T) * scale, f16 into LDS
  int qrow = min(q0 + r, 1024);
  v16h a0 = ld_a(qb + (size_t)qrow * TS + h * 16);
  v16h a1 = ld_a(qb + (size_t)qrow * TS + h * 16 + 32);
  for (int nt = wave; nt < 65; nt += 4) {
    int n0 = nt * 16;
    int krow = min(n0 + r, 1024);
    const half_t* kp = kb_ + (size_t)krow * TS + h * 16;
    v8f acc = {};
    acc = wmma32(a0, ld_b(kp), acc);
    acc = wmma32(a1, ld_b(kp + 32), acc);
    int col = n0 + r;
    union { float f[8]; v8f v; } cu; cu.v = acc;
    for (int e = 0; e < 8; ++e) {
      float sv = (col < 1025) ? cu.f[e] * 0.125f : -60000.f;  // 1/sqrt(64)
      S[e + 8 * h][col] = (half_t)sv;
    }
  }
  __syncthreads();

  // Phase 2: row softmax (fp32 math, f16 storage). 8 parts x 132 cols.
  int row = tid & 15, part = tid >> 4;
  int c0 = part * 132;
  float mx = -1e30f;
  for (int c = c0; c < c0 + 132; ++c) mx = fmaxf(mx, (float)S[row][c]);
  red[row][part] = mx;
  __syncthreads();
  if (tid < 16) { float m = red[tid][0]; for (int j = 1; j < 8; ++j) m = fmaxf(m, red[tid][j]); rowstat[tid] = m; }
  __syncthreads();
  float rmax = rowstat[row];
  float sum = 0.f;
  for (int c = c0; c < c0 + 132; ++c) {
    float e = __expf((float)S[row][c] - rmax);
    sum += e;
    S[row][c] = (half_t)e;
  }
  red[row][part] = sum;
  __syncthreads();
  if (tid < 16) { float t = 0.f; for (int j = 0; j < 8; ++j) t += red[tid][j]; rowstat[tid] = 1.f / t; }
  __syncthreads();
  float rinv = rowstat[row];
  for (int c = c0; c < c0 + 132; ++c) S[row][c] = (half_t)((float)S[row][c] * rinv);
  __syncthreads();

  // Phase 3: O = P @ V. Wave owns a 16-wide d-tile; K loop covers 1056 cols.
  int d0 = wave * 16;
  int d = d0 + r;
  v8f acc = {};
  for (int kt = 0; kt < 33; ++kt) {
    int kbb = kt * 32;
    v16h pa = ld_a(&S[r][kbb + h * 16]);
    union { _Float16 e[16]; v16h v; } bu;
    for (int e = 0; e < 16; ++e) {
      int kk = kbb + h * 16 + e;
      int tok = min(kk, 1024);   // padded rows have P==0
      bu.e[e] = vb[(size_t)tok * TS + d];
    }
    acc = wmma32(pa, bu.v, acc);
  }
  union { float f[8]; v8f v; } cu; cu.v = acc;
  for (int e = 0; e < 8; ++e) {
    int m = q0 + e + 8 * h;
    if (m < 1025)
      o[((size_t)b * 1025 + m) * 768 + hd * 64 + d] = (half_t)cu.f[e];
  }
}

// ---------------------------------------------------------------------------
__global__ __launch_bounds__(256) void pool_kernel(const float* __restrict__ x,
                                                   float* __restrict__ maxp,
                                                   float* __restrict__ avgp) {
  int b = blockIdx.x;
  int c = blockIdx.y * 256 + threadIdx.x;
  const float* xb = x + ((size_t)b * 1025 + 1) * 768 + c;
  float mx = -1e30f, s = 0.f;
  for (int t = 0; t < 1024; ++t) {
    float v = xb[(size_t)t * 768];
    mx = fmaxf(mx, v);
    s += v;
  }
  maxp[b * 768 + c] = mx;
  avgp[b * 768 + c] = s * (1.f / 1024.f);
}

__global__ __launch_bounds__(256) void gate_kernel(const float* __restrict__ maxp,
                                                   const float* __restrict__ avgp,
                                                   const float* __restrict__ ca1,
                                                   const float* __restrict__ ca2,
                                                   float* __restrict__ gate) {
  __shared__ float t1[192], t2[192];
  int b = blockIdx.x, tid = threadIdx.x;
  if (tid < 192) {
    float s1 = 0.f, s2 = 0.f;
    for (int i = 0; i < 768; ++i) {
      float w = ca1[(size_t)i * 192 + tid];
      s1 += maxp[b * 768 + i] * w;
      s2 += avgp[b * 768 + i] * w;
    }
    t1[tid] = fmaxf(s1, 0.f);
    t2[tid] = fmaxf(s2, 0.f);
  }
  __syncthreads();
  for (int c = tid; c < 768; c += 256) {
    float g = 0.f;
    for (int j = 0; j < 192; ++j) g += (t1[j] + t2[j]) * ca2[(size_t)j * 768 + c];
    gate[b * 768 + c] = 1.f / (1.f + __expf(-g));
  }
}

__global__ __launch_bounds__(256) void scale_kernel(float* __restrict__ out,
                                                    const float* __restrict__ gate) {
  size_t idx = (size_t)blockIdx.x * 256 + threadIdx.x;
  if (idx >= (size_t)8200 * 768) return;
  size_t m = idx / 768;
  int c = (int)(idx % 768);
  int b = (int)(m / 1025);
  int tok = (int)(m % 1025);
  if (tok != 0) out[idx] *= (1.f + gate[b * 768 + c]);
}

// ---------------------------------------------------------------------------
extern "C" void kernel_launch(void* const* d_in, const int* in_sizes, int n_in,
                              void* d_out, int out_size, void* d_ws, size_t ws_size,
                              hipStream_t stream) {
  (void)in_sizes; (void)n_in; (void)out_size; (void)ws_size;
  const float* x      = (const float*)d_in[0];
  const float* ln1_w  = (const float*)d_in[1];
  const float* ln1_b  = (const float*)d_in[2];
  const float* qkv_w  = (const float*)d_in[3];
  const float* proj_w = (const float*)d_in[4];
  const float* proj_b = (const float*)d_in[5];
  const float* ln2_w  = (const float*)d_in[6];
  const float* ln2_b  = (const float*)d_in[7];
  const float* fc1_w  = (const float*)d_in[8];
  const float* fc1_b  = (const float*)d_in[9];
  const float* fc2_w  = (const float*)d_in[10];
  const float* fc2_b  = (const float*)d_in[11];
  const float* ca1_w  = (const float*)d_in[12];
  const float* ca2_w  = (const float*)d_in[13];
  float* out = (float*)d_out;

  const int M = 8200, E = 768, H3 = 2304, HID = 3072;
  char* ws = (char*)d_ws;
  size_t off = 0;
  auto alloc = [&](size_t bytes) {
    char* p = ws + off;
    off = (off + bytes + 255) & ~(size_t)255;
    return p;
  };
  half_t* wq_t = (half_t*)alloc((size_t)H3 * E * 2);
  half_t* wp_t = (half_t*)alloc((size_t)E * E * 2);
  half_t* w1_t = (half_t*)alloc((size_t)HID * E * 2);
  half_t* w2_t = (half_t*)alloc((size_t)E * HID * 2);
  half_t* h1   = (half_t*)alloc((size_t)M * E * 2);
  half_t* qkv  = (half_t*)alloc((size_t)M * H3 * 2);
  half_t* ob   = (half_t*)alloc((size_t)M * E * 2);
  float*  x1   = (float*)alloc((size_t)M * E * 4);
  half_t* h2   = (half_t*)alloc((size_t)M * E * 2);
  half_t* ffn  = (half_t*)alloc((size_t)M * HID * 2);
  float*  maxp = (float*)alloc((size_t)8 * E * 4);
  float*  avgp = (float*)alloc((size_t)8 * E * 4);
  float*  gate = (float*)alloc((size_t)8 * E * 4);

  auto cdiv = [](int a, int b) { return (a + b - 1) / b; };

  wcvt_kernel<<<cdiv(E * H3, 256), 256, 0, stream>>>(qkv_w, wq_t, E, H3);
  wcvt_kernel<<<cdiv(E * E, 256), 256, 0, stream>>>(proj_w, wp_t, E, E);
  wcvt_kernel<<<cdiv(E * HID, 256), 256, 0, stream>>>(fc1_w, w1_t, E, HID);
  wcvt_kernel<<<cdiv(HID * E, 256), 256, 0, stream>>>(fc2_w, w2_t, HID, E);

  ln_kernel<<<M, 256, 0, stream>>>(x, ln1_w, ln1_b, h1);
  gemm_kernel<<<dim3(H3 / 64, cdiv(M, 64)), 128, 0, stream>>>(
      h1, wq_t, nullptr, nullptr, qkv, nullptr, M, H3, E, 0);
  attn_kernel<<<dim3(65, 12, 8), 128, 0, stream>>>(qkv, ob);
  gemm_kernel<<<dim3(E / 64, cdiv(M, 64)), 128, 0, stream>>>(
      ob, wp_t, proj_b, x, nullptr, x1, M, E, E, 2);
  ln_kernel<<<M, 256, 0, stream>>>(x1, ln2_w, ln2_b, h2);
  gemm_kernel<<<dim3(HID / 64, cdiv(M, 64)), 128, 0, stream>>>(
      h2, w1_t, fc1_b, nullptr, ffn, nullptr, M, HID, E, 1);
  gemm_kernel<<<dim3(E / 64, cdiv(M, 64)), 128, 0, stream>>>(
      ffn, w2_t, fc2_b, x1, nullptr, out, M, E, HID, 2);
  pool_kernel<<<dim3(8, 3), 256, 0, stream>>>(out, maxp, avgp);
  gate_kernel<<<8, 256, 0, stream>>>(maxp, avgp, ca1_w, ca2_w, gate);
  scale_kernel<<<cdiv(M * E, 256), 256, 0, stream>>>(out, gate);
}
